// Seq2SeqBridge_47416438948172
// MI455X (gfx1250) — matrix operand order repaired and checked by reading, not compiled, not run on hardware
//
#include <hip/hip_runtime.h>
#include <hip/hip_bf16.h>
#include <stdint.h>

typedef __attribute__((ext_vector_type(16))) _Float16 v16h;
typedef __attribute__((ext_vector_type(8)))  float    v8f;

// ---------------- fixed problem dims (from setup_inputs) ----------------
// B=32, E=H=512 (decoder), 4H=2048 gates, T=max_len=512
// encoder: Hd=256 per dir, 4Hd=1024 gates, K = 512(x) + 256(h) = 768

// ---------------- workspace layout (byte offsets, 256B aligned) ---------
static constexpr size_t alup(size_t x) { return (x + 255) & ~(size_t)255; }
static constexpr size_t SZ_BDEC = (size_t)2048 * 512 * 2;   // packed f16 dec matrix
static constexpr size_t SZ_BENC = (size_t)1024 * 768 * 2;   // packed f16 enc matrix (stride S)
static constexpr size_t OFF_BHH0   = 0;
static constexpr size_t OFF_BHH1   = OFF_BHH0 + alup(SZ_BDEC);
static constexpr size_t OFF_BIH1   = OFF_BHH1 + alup(SZ_BDEC);
static constexpr size_t OFF_BEF0   = OFF_BIH1 + alup(SZ_BDEC);   // + (layer*2+dir)*SZ_BENC
static constexpr size_t OFF_CONST0 = OFF_BEF0 + 4 * alup(SZ_BENC);
static constexpr size_t OFF_BSUM1  = OFF_CONST0 + alup(2048 * 4);
static constexpr size_t OFF_BSE    = OFF_BSUM1 + alup(2048 * 4); // + (layer*2+dir)*4096
static constexpr size_t OFF_H0A    = OFF_BSE + 4 * alup(1024 * 4);
// decoder state strides: 32*512*4 = 65536 (aligned): H0A,H0B,H1A,H1B,C0,C1
static constexpr size_t DSTRIDE    = (size_t)32 * 512 * 4;
static constexpr size_t OFF_H1A    = OFF_H0A + 2 * DSTRIDE;
static constexpr size_t OFF_C0     = OFF_H0A + 4 * DSTRIDE;
static constexpr size_t OFF_C1     = OFF_H0A + 5 * DSTRIDE;
static constexpr size_t OFF_P1     = OFF_H0A + 6 * DSTRIDE;      // 32x2048 f32
// encoder state strides: 32*256*4 = 32768 (aligned): HFA,HFB,HBA,HBB,CF,CB
static constexpr size_t ESTRIDE    = (size_t)32 * 256 * 4;
static constexpr size_t OFF_HFA    = OFF_P1 + alup((size_t)32 * 2048 * 4);
static constexpr size_t OFF_CF     = OFF_HFA + 4 * ESTRIDE;
static constexpr size_t OFF_BAR    = OFF_HFA + 6 * ESTRIDE;      // barrier slot (256B)
static constexpr size_t OFF_RES    = OFF_BAR + 256;              // [512,32,512] f32
static constexpr size_t OFF_Y0     = OFF_RES + alup((size_t)512 * 32 * 512 * 4);

// d_out float offsets: enc_outs_final [32,512,512] | dec_h [2,32,512] | dec_c [2,32,512]
static constexpr size_t OUT_H = (size_t)32 * 512 * 512;
static constexpr size_t OUT_C = OUT_H + (size_t)2 * 32 * 512;

// ---------------- device helpers ----------------
__device__ __forceinline__ float sigf(float x) { return 1.0f / (1.0f + __expf(-x)); }

// grid-wide barrier (persistent kernel, all blocks resident)
__device__ void grid_barrier(unsigned* cnt, unsigned* gen, unsigned nb) {
  __syncthreads();
  if (threadIdx.x == 0) {
    __threadfence();
    volatile unsigned* vgen = (volatile unsigned*)gen;
    unsigned g = *vgen;
    unsigned prev = atomicAdd(cnt, 1u);
    if (prev == nb - 1u) {
      *((volatile unsigned*)cnt) = 0u;
      __threadfence();
      atomicAdd(gen, 1u);
    } else {
      while (*vgen == g) { __builtin_amdgcn_s_sleep(2); }
    }
    __threadfence();
  }
  __syncthreads();
}

// Stage 16 rows [r0, r0+16) x K cols of a row-major f32 matrix into LDS as f16,
// pre-swizzled into a WMMA A-fragment layout with contiguous k order (matches B packing):
//   frag(kt): lane L (grp=L>>4, m=L&15) holds halves h=0..15 -> k=kt*32+grp*16+h
// so the per-wave fragment load is a contiguous 32B ds_load.
__device__ __forceinline__ void stage_A16(_Float16* sm, const float* __restrict__ src,
                                          int ld, int r0, int K) {
  const int half_k = K >> 1;
  for (int p = threadIdx.x; p < 16 * half_k; p += 256) {
    const int mrow = p / half_k;
    const int kp = (p - mrow * half_k) * 2;          // even k
    const float2 v = *(const float2*)(src + (size_t)(r0 + mrow) * ld + kp);
    const int kt = kp >> 5;
    const int kl = kp & 31;
    const int lane = ((kl >> 4) & 1) * 16 + mrow;
    const int h = kl & 15;
    _Float16* d = sm + ((size_t)kt * 32 + lane) * 16 + h;
    d[0] = (_Float16)v.x;
    d[1] = (_Float16)v.y;
  }
}

// Accumulate the 4 gate tiles (i,f,g,o) for column tile jt from LDS-staged A.
// Bp pre-packed: [(gi*NTG + jt)*KTS + kt][lane][16 halves]
template <int KTS, int NTG>
__device__ __forceinline__ void gemm_gates_lds(v8f* acc, const _Float16* sm,
                                               const _Float16* __restrict__ Bp,
                                               int jt, int lane) {
#pragma unroll
  for (int kt = 0; kt < KTS; ++kt) {
    const v16h a = *(const v16h*)(sm + ((size_t)kt * 32 + lane) * 16);
    if (kt + 1 < KTS)
      __builtin_prefetch(Bp + (((size_t)jt * KTS + kt + 1) * 32 + lane) * 16, 0, 3);
#pragma unroll
    for (int gi = 0; gi < 4; ++gi) {
      const v16h b = *(const v16h*)(Bp + (((size_t)(gi * NTG + jt) * KTS + kt) * 32 + lane) * 16);
      acc[gi] = __builtin_amdgcn_wmma_f32_16x16x32_f16(false, a, false, b, (short)0, acc[gi], false, false);
    }
  }
}

__device__ __forceinline__ void zero_acc(v8f* acc) {
#pragma unroll
  for (int gi = 0; gi < 4; ++gi) {
    v8f z = {0.f, 0.f, 0.f, 0.f, 0.f, 0.f, 0.f, 0.f};
    acc[gi] = z;
  }
}

// ---------------- prep kernels ----------------
__global__ void init_bar_kernel(unsigned* bar) {
  if (threadIdx.x < 64) bar[threadIdx.x] = 0u;
}

// Repack W [N x K] (row-major f32) into WMMA-B f16 fragments:
// packed[((nt*KT + kt)*32 + lane)*16 + h] = W[nt*16 + (lane&15), kt*32 + (lane>>4)*16 + h]
__global__ void pack_w_kernel(const float* __restrict__ W, _Float16* __restrict__ out, int N, int K) {
  const int KT = K / 32;
  const size_t total = (size_t)N * K;
  for (size_t idx = (size_t)blockIdx.x * blockDim.x + threadIdx.x; idx < total;
       idx += (size_t)gridDim.x * blockDim.x) {
    int h = (int)(idx & 15);
    int L = (int)((idx >> 4) & 31);
    size_t rest = idx >> 9;
    int kt = (int)(rest % KT);
    int nt = (int)(rest / KT);
    int n = nt * 16 + (L & 15);
    int k = kt * 32 + (L >> 4) * 16 + h;
    out[idx] = (_Float16)W[(size_t)n * K + k];
  }
}

// Encoder: fuse Wih [1024x512] and Whh [1024x256] into one K=768 packed matrix.
__global__ void pack_enc_kernel(const float* __restrict__ Wih, const float* __restrict__ Whh,
                                _Float16* __restrict__ out) {
  const int K = 768, KT = 24, N = 1024;
  const size_t total = (size_t)N * K;
  for (size_t idx = (size_t)blockIdx.x * blockDim.x + threadIdx.x; idx < total;
       idx += (size_t)gridDim.x * blockDim.x) {
    int h = (int)(idx & 15);
    int L = (int)((idx >> 4) & 31);
    size_t rest = idx >> 9;
    int kt = (int)(rest % KT);
    int nt = (int)(rest / KT);
    int n = nt * 16 + (L & 15);
    int k = kt * 32 + (L >> 4) * 16 + h;
    float w = (k < 512) ? Wih[(size_t)n * 512 + k] : Whh[(size_t)n * 256 + (k - 512)];
    out[idx] = (_Float16)w;
  }
}

// const0[n] = dot(Wih0[n,:], emb[BOS]) + bih0[n] + bhh0[n]   (decoder input is constant)
__global__ void dec_const_kernel(const float* __restrict__ Wih0, const float* __restrict__ bih0,
                                 const float* __restrict__ bhh0, const float* __restrict__ x0,
                                 float* __restrict__ c0) {
  int n = blockIdx.x * blockDim.x + threadIdx.x;
  if (n < 2048) {
    float s = bih0[n] + bhh0[n];
    const float* wr = Wih0 + (size_t)n * 512;
    for (int k = 0; k < 512; ++k) s += wr[k] * x0[k];
    c0[n] = s;
  }
}

__global__ void vec_add_kernel(const float* __restrict__ a, const float* __restrict__ b,
                               float* __restrict__ o, int n) {
  int i = blockIdx.x * blockDim.x + threadIdx.x;
  if (i < n) o[i] = a[i] + b[i];
}

// ---------------- decoder: persistent, 512 steps, 2 grid barriers/step ----------------
// 16 blocks x 256 threads. Blocks 0..7: layer-0 tasks; 8..15: layer-1 recurrent part.
// Per block all 8 waves share m-tile (mt = (bb>>2)&1), so A is staged once into LDS.
__global__ void __launch_bounds__(256) dec_kernel(const float* __restrict__ enc_h,
                                                  const float* __restrict__ enc_c,
                                                  char* __restrict__ ws) {
  __shared__ __align__(32) _Float16 smA[16 * 512];   // 16 KB: one 16x512 A panel, f16 swizzled

  const float* cst0 = (const float*)(ws + OFF_CONST0);
  const float* bs1  = (const float*)(ws + OFF_BSUM1);
  float* c0 = (float*)(ws + OFF_C0);
  float* c1 = (float*)(ws + OFF_C1);
  float* P1 = (float*)(ws + OFF_P1);
  float* res = (float*)(ws + OFF_RES);
  unsigned* bar = (unsigned*)(ws + OFF_BAR);

  const int tid = blockIdx.x * 256 + threadIdx.x;
  const int stride = gridDim.x * 256;
  for (int i = tid; i < 32 * 512; i += stride) {  // carry init = (enc_h, enc_c)
    ((float*)(ws + OFF_H0A))[i] = enc_h[i];
    ((float*)(ws + OFF_H1A))[i] = enc_h[32 * 512 + i];
    c0[i] = enc_c[i];
    c1[i] = enc_c[32 * 512 + i];
  }
  grid_barrier(bar, bar + 1, gridDim.x);

  const int bb   = blockIdx.x;
  const int w    = threadIdx.x >> 5;   // wave in block
  const int lane = threadIdx.x & 31;
  const int nl   = lane & 15;
  const int grp  = lane >> 4;
  const int mtb  = (bb >> 2) & 1;      // A row-tile, constant per block
  const int jt   = (bb & 3) * 8 + w;   // column tile 0..31
  const bool l0side = (bb < 8);

  for (int t = 0; t < 512; ++t) {
    const size_t cur = (size_t)(t & 1) * DSTRIDE;
    const size_t nxt = (size_t)((t & 1) ^ 1) * DSTRIDE;
    float* h0n = (float*)(ws + OFF_H0A + nxt);
    float* h1n = (float*)(ws + OFF_H1A + nxt);

    // ---- stage A: layer-0 gates (blocks 0..7) | P1 = h1@Whh1^T + b (blocks 8..15) ----
    {
      const size_t srcOff = (l0side ? OFF_H0A : OFF_H1A) + cur;
      stage_A16(smA, (const float*)(ws + srcOff), 512, mtb * 16, 512);
      __syncthreads();
      const _Float16* Bp = (const _Float16*)(ws + (l0side ? OFF_BHH0 : OFF_BHH1));
      v8f acc[4]; zero_acc(acc);
      gemm_gates_lds<16, 32>(acc, smA, Bp, jt, lane);
      if (l0side) {
        const float bi = cst0[0 * 512 + jt * 16 + nl];
        const float bf = cst0[1 * 512 + jt * 16 + nl];
        const float bg = cst0[2 * 512 + jt * 16 + nl];
        const float bo = cst0[3 * 512 + jt * 16 + nl];
#pragma unroll
        for (int v = 0; v < 8; ++v) {
          const int m = mtb * 16 + grp * 8 + v;
          const int j = jt * 16 + nl;
          float iv = sigf(acc[0][v] + bi);
          float fv = sigf(acc[1][v] + bf);
          float gv = tanhf(acc[2][v] + bg);
          float ov = sigf(acc[3][v] + bo);
          float cc = fv * c0[m * 512 + j] + iv * gv;
          c0[m * 512 + j] = cc;
          h0n[m * 512 + j] = ov * tanhf(cc);
        }
      } else {
        const float bi = bs1[0 * 512 + jt * 16 + nl];
        const float bf = bs1[1 * 512 + jt * 16 + nl];
        const float bg = bs1[2 * 512 + jt * 16 + nl];
        const float bo = bs1[3 * 512 + jt * 16 + nl];
#pragma unroll
        for (int v = 0; v < 8; ++v) {
          const int m = mtb * 16 + grp * 8 + v;
          const int j = jt * 16 + nl;
          P1[m * 2048 + 0 * 512 + j] = acc[0][v] + bi;
          P1[m * 2048 + 1 * 512 + j] = acc[1][v] + bf;
          P1[m * 2048 + 2 * 512 + j] = acc[2][v] + bg;
          P1[m * 2048 + 3 * 512 + j] = acc[3][v] + bo;
        }
      }
    }
    grid_barrier(bar, bar + 1, gridDim.x);

    // ---- stage B: layer-1 gates = h0_new@Wih1^T + P1 ; update (h1,c1); emit result[t] ----
    if (l0side) {
      stage_A16(smA, (const float*)(ws + OFF_H0A + nxt), 512, mtb * 16, 512);
      __syncthreads();
      v8f acc[4]; zero_acc(acc);
      gemm_gates_lds<16, 32>(acc, smA, (const _Float16*)(ws + OFF_BIH1), jt, lane);
#pragma unroll
      for (int v = 0; v < 8; ++v) {
        const int m = mtb * 16 + grp * 8 + v;
        const int j = jt * 16 + nl;
        float iv = sigf(acc[0][v] + P1[m * 2048 + j]);
        float fv = sigf(acc[1][v] + P1[m * 2048 + 512 + j]);
        float gv = tanhf(acc[2][v] + P1[m * 2048 + 1024 + j]);
        float ov = sigf(acc[3][v] + P1[m * 2048 + 1536 + j]);
        float cc = fv * c1[m * 512 + j] + iv * gv;
        c1[m * 512 + j] = cc;
        float hv = ov * tanhf(cc);
        h1n[m * 512 + j] = hv;
        res[(size_t)t * 32 * 512 + m * 512 + j] = hv;
      }
    }
    grid_barrier(bar, bar + 1, gridDim.x);
  }
}

// ---------------- encoder layer: persistent, fwd+bwd concurrent, 1 barrier/step ---------
// 8 blocks x 256 threads; block encodes (dir, mt): dir = bb>>2, mt = (bb>>1)&1,
// jt = (bb&1)*8 + wave. Single K=768 GEMM per step per dir: A = [x_t | h_dir] staged in LDS.
__global__ void __launch_bounds__(256) enc_kernel(char* __restrict__ ws,
                                                  float* __restrict__ dout, int layer) {
  __shared__ __align__(32) _Float16 smE[16 * 768];   // 24 KB: [x_t | h] panel, f16 swizzled

  float* y0 = (float*)(ws + OFF_Y0);
  unsigned* bar = (unsigned*)(ws + OFF_BAR);

  const int tid = blockIdx.x * 256 + threadIdx.x;
  const int stride = gridDim.x * 256;
  for (int i = tid; i < 32 * 256; i += stride) {  // zero initial states (all buffers)
    ((float*)(ws + OFF_HFA + 0 * ESTRIDE))[i] = 0.f;
    ((float*)(ws + OFF_HFA + 1 * ESTRIDE))[i] = 0.f;
    ((float*)(ws + OFF_HFA + 2 * ESTRIDE))[i] = 0.f;
    ((float*)(ws + OFF_HFA + 3 * ESTRIDE))[i] = 0.f;
    ((float*)(ws + OFF_CF + 0 * ESTRIDE))[i] = 0.f;
    ((float*)(ws + OFF_CF + 1 * ESTRIDE))[i] = 0.f;
  }
  grid_barrier(bar, bar + 1, gridDim.x);

  const int bb   = blockIdx.x;
  const int w    = threadIdx.x >> 5;
  const int lane = threadIdx.x & 31;
  const int nl   = lane & 15;
  const int grp  = lane >> 4;
  const int dir  = bb >> 2;            // 0 = fwd, 1 = bwd
  const int mtb  = (bb >> 1) & 1;
  const int jt   = (bb & 1) * 8 + w;   // 0..15

  const size_t xOff = layer ? OFF_Y0 : OFF_RES;
  const _Float16* Bp = (const _Float16*)(ws + OFF_BEF0 + ((size_t)layer * 2 + dir) * alup(SZ_BENC));
  const float* bs = (const float*)(ws + OFF_BSE + ((size_t)layer * 2 + dir) * 4096);
  float* cb = (float*)(ws + OFF_CF + (size_t)dir * ESTRIDE);
  float* dh = dout + OUT_H + (size_t)layer * 32 * 512;
  float* dc = dout + OUT_C + (size_t)layer * 32 * 512;

  for (int s = 0; s < 512; ++s) {
    const int tx = dir ? (511 - s) : s;
    const float* hc = (const float*)(ws + OFF_HFA + ((size_t)dir * 2 + (s & 1)) * ESTRIDE);
    float* hn = (float*)(ws + OFF_HFA + ((size_t)dir * 2 + ((s & 1) ^ 1)) * ESTRIDE);
    const float* xt = (const float*)(ws + xOff) + (size_t)tx * 32 * 512;

    stage_A16(smE, xt, 512, mtb * 16, 512);            // fragments kt 0..15  (x part)
    stage_A16(smE + 16 * 512, hc, 256, mtb * 16, 256); // fragments kt 16..23 (h part)
    __syncthreads();

    v8f acc[4]; zero_acc(acc);
    gemm_gates_lds<24, 16>(acc, smE, Bp, jt, lane);

    const float bi = bs[0 * 256 + jt * 16 + nl];
    const float bf = bs[1 * 256 + jt * 16 + nl];
    const float bg = bs[2 * 256 + jt * 16 + nl];
    const float bo = bs[3 * 256 + jt * 16 + nl];
#pragma unroll
    for (int v = 0; v < 8; ++v) {
      const int m = mtb * 16 + grp * 8 + v;
      const int j = jt * 16 + nl;
      float iv = sigf(acc[0][v] + bi);
      float fv = sigf(acc[1][v] + bf);
      float gv = tanhf(acc[2][v] + bg);
      float ov = sigf(acc[3][v] + bo);
      float cc = fv * cb[m * 256 + j] + iv * gv;
      cb[m * 256 + j] = cc;
      float hv = ov * tanhf(cc);
      hn[m * 256 + j] = hv;
      if (layer == 0) {
        y0[(size_t)tx * 32 * 512 + m * 512 + dir * 256 + j] = hv;             // [T,B,2Hd]
      } else {
        dout[(size_t)m * 512 * 512 + (size_t)tx * 512 + dir * 256 + j] = hv;  // [B,T,2Hd]
      }
      if (s == 511) {  // final carry (fwd after t=511, bwd after t=0)
        dh[m * 512 + dir * 256 + j] = hv;
        dc[m * 512 + dir * 256 + j] = cc;
      }
    }
    grid_barrier(bar, bar + 1, gridDim.x);
  }
}

// ---------------- launch ----------------
extern "C" void kernel_launch(void* const* d_in, const int* in_sizes, int n_in,
                              void* d_out, int out_size, void* d_ws, size_t ws_size,
                              hipStream_t stream) {
  (void)in_sizes; (void)n_in; (void)out_size; (void)ws_size;
  const float* enc_h   = (const float*)d_in[1];
  const float* enc_c   = (const float*)d_in[2];
  const float* dec_emb = (const float*)d_in[3];
  const float* dec_Wih = (const float*)d_in[4];
  const float* dec_Whh = (const float*)d_in[5];
  const float* dec_bih = (const float*)d_in[6];
  const float* dec_bhh = (const float*)d_in[7];
  const float* eWf  = (const float*)d_in[8];
  const float* eUf  = (const float*)d_in[9];
  const float* ebif = (const float*)d_in[10];
  const float* ebhf = (const float*)d_in[11];
  const float* eWb  = (const float*)d_in[12];
  const float* eUb  = (const float*)d_in[13];
  const float* ebib = (const float*)d_in[14];
  const float* ebhb = (const float*)d_in[15];
  char* ws = (char*)d_ws;
  float* out = (float*)d_out;

  init_bar_kernel<<<1, 64, 0, stream>>>((unsigned*)(ws + OFF_BAR));

  // pack weights into WMMA-B f16 fragment layout (L2-resident thereafter)
  pack_w_kernel<<<1024, 256, 0, stream>>>(dec_Whh,              (_Float16*)(ws + OFF_BHH0), 2048, 512);
  pack_w_kernel<<<1024, 256, 0, stream>>>(dec_Whh + 2048 * 512, (_Float16*)(ws + OFF_BHH1), 2048, 512);
  pack_w_kernel<<<1024, 256, 0, stream>>>(dec_Wih + 2048 * 512, (_Float16*)(ws + OFF_BIH1), 2048, 512);
  pack_enc_kernel<<<768, 256, 0, stream>>>(eWf,              eUf,              (_Float16*)(ws + OFF_BEF0 + 0 * alup(SZ_BENC)));
  pack_enc_kernel<<<768, 256, 0, stream>>>(eWb,              eUb,              (_Float16*)(ws + OFF_BEF0 + 1 * alup(SZ_BENC)));
  pack_enc_kernel<<<768, 256, 0, stream>>>(eWf + 1024 * 512, eUf + 1024 * 256, (_Float16*)(ws + OFF_BEF0 + 2 * alup(SZ_BENC)));
  pack_enc_kernel<<<768, 256, 0, stream>>>(eWb + 1024 * 512, eUb + 1024 * 256, (_Float16*)(ws + OFF_BEF0 + 3 * alup(SZ_BENC)));

  // fold constant decoder input projection + biases; plain bias sums elsewhere
  dec_const_kernel<<<8, 256, 0, stream>>>(dec_Wih, dec_bih, dec_bhh,
                                          dec_emb + 512 /* BOS_TOK_IDX=1 */,
                                          (float*)(ws + OFF_CONST0));
  vec_add_kernel<<<8, 256, 0, stream>>>(dec_bih + 2048, dec_bhh + 2048, (float*)(ws + OFF_BSUM1), 2048);
  vec_add_kernel<<<4, 256, 0, stream>>>(ebif,        ebhf,        (float*)(ws + OFF_BSE + 0 * 4096), 1024);
  vec_add_kernel<<<4, 256, 0, stream>>>(ebib,        ebhb,        (float*)(ws + OFF_BSE + 1 * 4096), 1024);
  vec_add_kernel<<<4, 256, 0, stream>>>(ebif + 1024, ebhf + 1024, (float*)(ws + OFF_BSE + 2 * 4096), 1024);
  vec_add_kernel<<<4, 256, 0, stream>>>(ebib + 1024, ebhb + 1024, (float*)(ws + OFF_BSE + 3 * 4096), 1024);

  // 512-step decoder, then 2 bidirectional encoder layers over its output
  dec_kernel<<<16, 256, 0, stream>>>(enc_h, enc_c, ws);
  enc_kernel<<<8, 256, 0, stream>>>(ws, out, 0);
  enc_kernel<<<8, 256, 0, stream>>>(ws, out, 1);
}
